// TransformerSquared_2808908611957
// MI455X (gfx1250) — matrix operand order repaired and checked by reading, not compiled, not run on hardware
//
#include <hip/hip_runtime.h>
#include <stdint.h>

// ---------------- problem constants ----------------
#define Lc 4
#define Bc 2
#define Sc 1024
#define Dc 1024
#define Hc 16
#define DHc 64
#define Rc 4
#define ROWSc (Bc*Sc)            // 2048
#define SCALEc 0.125f            // 1/sqrt(64)
#define EPSc 1e-5f

typedef __bf16 bf16_t;
typedef __attribute__((ext_vector_type(16))) __bf16 v16bf;
typedef __attribute__((ext_vector_type(8)))  float  v8f;

union AFrag { v16bf v; bf16_t e[16]; uint32_t u[8]; };
union CAcc  { v8f   v; float    f[8]; };

// CDNA5 async memory->LDS DMA (ASYNCcnt-tracked), per CDNA5 ISA.
// vdst = 32-bit LDS byte offset, vaddr = 64-bit global address, SADDR=off (GV mode).
#define ASYNC_LD_B128(ldsoff, gptr) \
  asm volatile("global_load_async_to_lds_b128 %0, %1, off" :: "v"(ldsoff), "v"(gptr) : "memory")
#define ASYNC_LD_B64(ldsoff, gptr) \
  asm volatile("global_load_async_to_lds_b64 %0, %1, off" :: "v"(ldsoff), "v"(gptr) : "memory")
#define WAIT_ASYNC() asm volatile("s_wait_asynccnt 0x0" ::: "memory")

// ---------------- generic batched WMMA bf16 GEMM ----------------
// C[z] = epilogue( alpha * A[z] x B[z] + bias ),  A: MxK row-major bf16,
// B: KxN row-major bf16 (or NxK row-major if transB). Batch offset for
// pointer P is (z/batchDiv)*sP1 + (z%batchDiv)*sP2 (elements).
struct GemmP {
  const bf16_t* A; long long lda, sA1, sA2;
  const bf16_t* B; long long ldb, sB1, sB2;
  float* Cf; bf16_t* Cbf; long long ldc, sC1, sC2;
  const float* bias;
  int M, N, K, batchDiv, transB, gelu, accumulate;
  float alpha;
};

#define LDS_PITCH 40                       // halves; 80B rows: 16B-aligned, 20-dword pitch
#define LDS_BUFB  (64 * LDS_PITCH * 2)     // bytes per buffer

__global__ __launch_bounds__(128)
void gemm_bf16_wmma(GemmP p) {
  const int z  = blockIdx.z;
  const int zb = z / p.batchDiv, zh = z % p.batchDiv;
  const bf16_t* A  = p.A + (long long)zb*p.sA1 + (long long)zh*p.sA2;
  const bf16_t* Bm = p.B + (long long)zb*p.sB1 + (long long)zh*p.sB2;
  const long long coff = (long long)zb*p.sC1 + (long long)zh*p.sC2;

  const int gm0 = blockIdx.y * 64;
  const int gn0 = blockIdx.x * 64;

  // double-buffered tiles; A: [m][k]; B: [k][n] (row-major) or [n][k] (transB)
  __shared__ bf16_t ldsA[2][64][LDS_PITCH];
  __shared__ bf16_t ldsB[2][64][LDS_PITCH];

  const int tid  = threadIdx.x;
  const int lane = tid & 31;
  const int wave = tid >> 5;
  const int wm = (wave >> 1) * 32;
  const int wn = (wave & 1)  * 32;
  const int fm   = lane & 15;
  const int half = lane >> 4;

  // ---- per-thread async staging descriptors (2x A B128, 2x B B128 / 4x B B64) ----
  const int rA0 = tid >> 2,          cA0 = (tid & 3) * 8;          // A chunk 0
  const int rA1 = (tid + 128) >> 2,  cA1 = ((tid + 128) & 3) * 8;  // A chunk 1
  const bf16_t* gA0 = A + (long long)(gm0 + rA0) * p.lda + cA0;
  const bf16_t* gA1 = A + (long long)(gm0 + rA1) * p.lda + cA1;
  const uint32_t lA0 = (uint32_t)(uintptr_t)&ldsA[0][rA0][cA0];
  const uint32_t lA1 = (uint32_t)(uintptr_t)&ldsA[0][rA1][cA1];

  // row-major B: 32 rows x 8 chunks of 8 halves
  const int rB0 = tid >> 3,          cB0 = (tid & 7) * 8;
  const int rB1 = (tid + 128) >> 3,  cB1 = ((tid + 128) & 7) * 8;
  const bf16_t* gB0 = Bm + (long long)rB0 * p.ldb + gn0 + cB0;
  const bf16_t* gB1 = Bm + (long long)rB1 * p.ldb + gn0 + cB1;
  const uint32_t lB0 = (uint32_t)(uintptr_t)&ldsB[0][rB0][cB0];
  const uint32_t lB1 = (uint32_t)(uintptr_t)&ldsB[0][rB1][cB1];

  // transposed B: 64 n-rows x 8 chunks of 4 halves along K
  int      rT[4]; int cT[4]; const bf16_t* gT[4]; uint32_t lT[4];
  #pragma unroll
  for (int t = 0; t < 4; ++t) {
    int idx = tid + t * 128;                 // 0..511
    rT[t] = idx >> 3; cT[t] = (idx & 7) * 4;
    gT[t] = Bm + (long long)(gn0 + rT[t]) * p.ldb + cT[t];
    lT[t] = (uint32_t)(uintptr_t)&ldsB[0][rT[t]][cT[t]];
  }

  auto stage = [&](int buf, int k0) {
    const uint32_t bo = (uint32_t)buf * LDS_BUFB;
    ASYNC_LD_B128(lA0 + bo, gA0 + k0);
    ASYNC_LD_B128(lA1 + bo, gA1 + k0);
    if (!p.transB) {
      const long long ko = (long long)k0 * p.ldb;
      ASYNC_LD_B128(lB0 + bo, gB0 + ko);
      ASYNC_LD_B128(lB1 + bo, gB1 + ko);
    } else {
      #pragma unroll
      for (int t = 0; t < 4; ++t) ASYNC_LD_B64(lT[t] + bo, gT[t] + k0);
    }
  };

  CAcc acc[2][2] = {};
  const int ksteps = p.K >> 5;

  stage(0, 0);
  for (int i = 0; i < ksteps; ++i) {
    WAIT_ASYNC();          // our waves' DMAs into buffer (i&1) are complete
    __syncthreads();       // every wave's DMAs visible; prev buffer fully consumed
    if (i + 1 < ksteps) stage((i + 1) & 1, (i + 1) << 5);

    const int b = i & 1;
    const bf16_t (*tA)[LDS_PITCH] = ldsA[b];
    const bf16_t (*tB)[LDS_PITCH] = ldsB[b];

    AFrag af[2], bfr[2];
    #pragma unroll
    for (int s = 0; s < 2; ++s) {
      const int mrow = wm + s * 16 + fm;
      #pragma unroll
      for (int j = 0; j < 8; ++j) {
        // A 16x32 frag: lanes0-15 K={0..7,16..23}, lanes16-31 K=+8
        int kk = ((j >> 2) << 4) + ((j & 3) << 1) + (half << 3);
        af[s].u[j] = *(const uint32_t*)&tA[mrow][kk];
      }
      const int ncol = wn + s * 16 + fm;
      #pragma unroll
      for (int j = 0; j < 8; ++j) {
        // B 32x16 frag: lanes0-15 K={0..7,16..23}, lanes16-31 K=+8
        int kk = ((j >> 2) << 4) + ((j & 3) << 1) + (half << 3);
        if (p.transB) {
          bfr[s].u[j] = *(const uint32_t*)&tB[ncol][kk];
        } else {
          bfr[s].e[2 * j]     = tB[kk][ncol];
          bfr[s].e[2 * j + 1] = tB[kk + 1][ncol];
        }
      }
    }
    #pragma unroll
    for (int mi = 0; mi < 2; ++mi)
      #pragma unroll
      for (int nj = 0; nj < 2; ++nj)
        acc[mi][nj].v = __builtin_amdgcn_wmma_f32_16x16x32_bf16(
            false, af[mi].v, false, bfr[nj].v, (short)0, acc[mi][nj].v, false, false);
  }

  // ---- epilogue: C/D layout (lane0-15: M=r, lane16-31: M=r+8; N=lane&15) ----
  #pragma unroll
  for (int mi = 0; mi < 2; ++mi)
    #pragma unroll
    for (int nj = 0; nj < 2; ++nj) {
      const int nglob = gn0 + wn + nj * 16 + fm;
      float bia = p.bias ? p.bias[nglob] : 0.0f;
      #pragma unroll
      for (int r = 0; r < 8; ++r) {
        int mglob = gm0 + wm + mi * 16 + r + half * 8;
        float v = acc[mi][nj].f[r] * p.alpha + bia;
        if (p.gelu) v = 0.5f * v * (1.0f + erff(v * 0.70710678118654752f));
        long long off = coff + (long long)mglob * p.ldc + nglob;
        if (p.Cf)  { if (p.accumulate) p.Cf[off] += v; else p.Cf[off] = v; }
        if (p.Cbf) p.Cbf[off] = (bf16_t)v;
      }
    }
}

// ---------------- LayerNorm (one block per row of D=1024) ----------------
__global__ __launch_bounds__(256)
void ln_kernel(const float* __restrict__ in, const float* __restrict__ g,
               const float* __restrict__ b, const float* __restrict__ adapt,
               float* outf, bf16_t* outbf, float* resid) {
  const long long row = blockIdx.x;
  const float* x = in + row * Dc;
  const int tid = threadIdx.x;
  float vals[4]; float s = 0.f, s2 = 0.f;
  #pragma unroll
  for (int t = 0; t < 4; ++t) { float v = x[tid + t * 256]; vals[t] = v; s += v; s2 += v * v; }
  __shared__ float r1[256], r2[256];
  r1[tid] = s; r2[tid] = s2; __syncthreads();
  for (int o = 128; o > 0; o >>= 1) {
    if (tid < o) { r1[tid] += r1[tid + o]; r2[tid] += r2[tid + o]; }
    __syncthreads();
  }
  float mean = r1[0] * (1.0f / Dc);
  float var  = r2[0] * (1.0f / Dc) - mean * mean;
  float inv  = rsqrtf(var + EPSc);
  #pragma unroll
  for (int t = 0; t < 4; ++t) {
    int c = tid + t * 256;
    float v = (vals[t] - mean) * inv * g[c] + b[c];
    if (adapt) v += adapt[c];
    long long off = row * Dc + c;
    if (outf)  outf[off]  = v;
    if (outbf) outbf[off] = (bf16_t)v;
    if (resid) resid[off] += v;
  }
}

// ------------- dual softmax difference: softmax(s1)-softmax(s2) -------------
__global__ __launch_bounds__(256)
void softmax_diff_kernel(const float* __restrict__ s1, const float* __restrict__ s2,
                         bf16_t* __restrict__ diff) {
  const long long row = blockIdx.x;            // B*H*S rows of length S
  const float* a = s1 + row * Sc;
  const float* b = s2 + row * Sc;
  const int tid = threadIdx.x;
  float va[4], vb[4], m1 = -3.4e38f, m2 = -3.4e38f;
  #pragma unroll
  for (int t = 0; t < 4; ++t) {
    va[t] = a[tid + t * 256]; vb[t] = b[tid + t * 256];
    m1 = fmaxf(m1, va[t]);    m2 = fmaxf(m2, vb[t]);
  }
  __shared__ float r1[256], r2[256];
  r1[tid] = m1; r2[tid] = m2; __syncthreads();
  for (int o = 128; o > 0; o >>= 1) {
    if (tid < o) { r1[tid] = fmaxf(r1[tid], r1[tid + o]); r2[tid] = fmaxf(r2[tid], r2[tid + o]); }
    __syncthreads();
  }
  m1 = r1[0]; m2 = r2[0]; __syncthreads();
  float sa = 0.f, sb = 0.f;
  #pragma unroll
  for (int t = 0; t < 4; ++t) {
    va[t] = expf(va[t] - m1); vb[t] = expf(vb[t] - m2);
    sa += va[t]; sb += vb[t];
  }
  r1[tid] = sa; r2[tid] = sb; __syncthreads();
  for (int o = 128; o > 0; o >>= 1) {
    if (tid < o) { r1[tid] += r1[tid + o]; r2[tid] += r2[tid + o]; }
    __syncthreads();
  }
  float ia = 1.0f / r1[0], ib = 1.0f / r2[0];
  #pragma unroll
  for (int t = 0; t < 4; ++t)
    diff[row * Sc + tid + t * 256] = (bf16_t)(va[t] * ia - vb[t] * ib);
}

// ---------------- small elementwise helpers ----------------
__global__ void f32_to_bf16_kernel(const float* __restrict__ in, bf16_t* __restrict__ out, long long n) {
  long long i = (long long)blockIdx.x * 256 + threadIdx.x;
  if (i < n) out[i] = (bf16_t)in[i];
}

__global__ void vavg_kernel(const bf16_t* __restrict__ qkv1, const bf16_t* __restrict__ qkv2,
                            bf16_t* __restrict__ vavg) {
  long long i = (long long)blockIdx.x * 256 + threadIdx.x;   // over ROWS*D
  long long r = i >> 10, c = i & 1023;
  long long src = r * (3 * Dc) + 2 * Dc + c;                 // v-part of qkv
  vavg[i] = (bf16_t)(0.5f * ((float)qkv1[src] + (float)qkv2[src]));
}

// adapt[l][i] = sigma[l] * sum_r U[l][i][r] * (sum_j V[l][r][j])
__global__ __launch_bounds__(256)
void adapt_kernel(const float* __restrict__ sigma, const float* __restrict__ U,
                  const float* __restrict__ V, float* __restrict__ adapt) {
  const int l = blockIdx.x;
  const float* Ul = U + (long long)l * Dc * Rc;
  const float* Vl = V + (long long)l * Rc * Dc;
  __shared__ float vs[Rc];
  __shared__ float red[256];
  const int tid = threadIdx.x;
  for (int r = 0; r < Rc; ++r) {
    float s = 0.f;
    for (int c = tid; c < Dc; c += 256) s += Vl[r * Dc + c];
    red[tid] = s; __syncthreads();
    for (int o = 128; o > 0; o >>= 1) { if (tid < o) red[tid] += red[tid + o]; __syncthreads(); }
    if (tid == 0) vs[r] = red[0];
    __syncthreads();
  }
  float sig = sigma[l];
  for (int c = tid; c < Dc; c += 256) {
    float s = 0.f;
    #pragma unroll
    for (int r = 0; r < Rc; ++r) s += Ul[c * Rc + r] * vs[r];
    adapt[(long long)l * Dc + c] = sig * s;
  }
}

// ---------------- host orchestration ----------------
static inline GemmP mkGemm(const bf16_t* A, long long lda, const bf16_t* B, long long ldb,
                           int M, int N, int K) {
  GemmP p{};
  p.A = A; p.lda = lda; p.B = B; p.ldb = ldb;
  p.M = M; p.N = N; p.K = K;
  p.batchDiv = 1; p.alpha = 1.0f; p.ldc = N;
  return p;
}

extern "C" void kernel_launch(void* const* d_in, const int* in_sizes, int n_in,
                              void* d_out, int out_size, void* d_ws, size_t ws_size,
                              hipStream_t stream) {
  const float* x     = (const float*)d_in[0];
  const float* sigma = (const float*)d_in[1];
  const float* U     = (const float*)d_in[2];
  const float* V     = (const float*)d_in[3];
  const float* Wqkv1 = (const float*)d_in[4];
  const float* Wqkv2 = (const float*)d_in[5];
  const float* Wp1   = (const float*)d_in[6];
  const float* bp1   = (const float*)d_in[7];
  const float* Wp2   = (const float*)d_in[8];
  const float* bp2   = (const float*)d_in[9];
  const float* Wo    = (const float*)d_in[10];
  const float* bo    = (const float*)d_in[11];
  const float* g_out = (const float*)d_in[12];
  const float* b_out = (const float*)d_in[13];
  const float* g1    = (const float*)d_in[14];
  const float* b1    = (const float*)d_in[15];
  const float* g2    = (const float*)d_in[16];
  const float* b2    = (const float*)d_in[17];
  const float* Wm1   = (const float*)d_in[18];
  const float* bm1   = (const float*)d_in[19];
  const float* Wm2   = (const float*)d_in[20];
  const float* bm2   = (const float*)d_in[21];
  const float* gf    = (const float*)d_in[22];
  const float* bfin  = (const float*)d_in[23];

  // ---- bump allocator over workspace ----
  char* ws = (char*)d_ws;
  size_t off = 0;
  auto alloc = [&](size_t bytes) -> void* {
    void* p = ws + off;
    off = (off + bytes + 255) & ~(size_t)255;
    return p;
  };
  const long long nWp   = (long long)Lc * (Dc/2) * Dc;
  const long long nWqkv = (long long)Lc * Dc * 3 * Dc;
  const long long nWo   = (long long)Lc * Dc * Dc;
  const long long nWm   = (long long)Lc * Dc * 4 * Dc;

  float*  adaptb = (float*)alloc((size_t)Lc * Dc * 4);
  bf16_t* wp1b   = (bf16_t*)alloc(nWp * 2);
  bf16_t* wp2b   = (bf16_t*)alloc(nWp * 2);
  bf16_t* wq1b   = (bf16_t*)alloc(nWqkv * 2);
  bf16_t* wq2b   = (bf16_t*)alloc(nWqkv * 2);
  bf16_t* wob    = (bf16_t*)alloc(nWo * 2);
  bf16_t* wm1b   = (bf16_t*)alloc(nWm * 2);
  bf16_t* wm2b   = (bf16_t*)alloc(nWm * 2);
  float*  xc     = (float*)alloc((size_t)ROWSc * Dc * 4);
  bf16_t* a_bf   = (bf16_t*)alloc((size_t)ROWSc * Dc * 2);
  bf16_t* x1p_bf = (bf16_t*)alloc((size_t)ROWSc * Dc * 2);
  bf16_t* x2p_bf = (bf16_t*)alloc((size_t)ROWSc * Dc * 2);
  bf16_t* qkv1_bf= (bf16_t*)alloc((size_t)ROWSc * 3 * Dc * 2);
  bf16_t* qkv2_bf= (bf16_t*)alloc((size_t)ROWSc * 3 * Dc * 2);
  bf16_t* vavg_bf= (bf16_t*)alloc((size_t)ROWSc * Dc * 2);
  float*  s1     = (float*)alloc((size_t)Bc * Hc * Sc * Sc * 4);
  float*  s2     = (float*)alloc((size_t)Bc * Hc * Sc * Sc * 4);
  bf16_t* diffb  = (bf16_t*)alloc((size_t)Bc * Hc * Sc * Sc * 2);
  bf16_t* ao_bf  = (bf16_t*)alloc((size_t)ROWSc * Dc * 2);
  float*  tmpf   = (float*)alloc((size_t)ROWSc * Dc * 4);
  bf16_t* m_bf   = (bf16_t*)alloc((size_t)ROWSc * 4 * Dc * 2);
  (void)ws_size; (void)in_sizes; (void)n_in; (void)out_size;

  auto cvt = [&](const float* src, bf16_t* dst, long long n) {
    f32_to_bf16_kernel<<<dim3((unsigned)((n + 255) / 256)), dim3(256), 0, stream>>>(src, dst, n);
  };
  cvt(Wp1,   wp1b, nWp);
  cvt(Wp2,   wp2b, nWp);
  cvt(Wqkv1, wq1b, nWqkv);
  cvt(Wqkv2, wq2b, nWqkv);
  cvt(Wo,    wob,  nWo);
  cvt(Wm1,   wm1b, nWm);
  cvt(Wm2,   wm2b, nWm);
  adapt_kernel<<<dim3(Lc), dim3(256), 0, stream>>>(sigma, U, V, adaptb);
  hipMemcpyAsync(xc, x, (size_t)ROWSc * Dc * 4, hipMemcpyDeviceToDevice, stream);

  auto gemm = [&](const GemmP& p, int batch) {
    dim3 grid((unsigned)(p.N / 64), (unsigned)(p.M / 64), (unsigned)batch);
    gemm_bf16_wmma<<<grid, dim3(128), 0, stream>>>(p);
  };

  for (int l = 0; l < Lc; ++l) {
    // a = LN(xc; g1,b1) + adapt[l]  (bf16)
    ln_kernel<<<dim3(ROWSc), dim3(256), 0, stream>>>(
        xc, g1 + l * Dc, b1 + l * Dc, adaptb + l * Dc, nullptr, a_bf, nullptr);

    // x1 = a[:, :512] @ Wp1 + bp1 ; x2 = a[:, 512:] @ Wp2 + bp2
    {
      GemmP p = mkGemm(a_bf, Dc, wp1b + (long long)l * (Dc/2) * Dc, Dc, ROWSc, Dc, Dc/2);
      p.bias = bp1 + l * Dc; p.Cbf = x1p_bf; gemm(p, 1);
      GemmP q = mkGemm(a_bf + Dc/2, Dc, wp2b + (long long)l * (Dc/2) * Dc, Dc, ROWSc, Dc, Dc/2);
      q.bias = bp2 + l * Dc; q.Cbf = x2p_bf; gemm(q, 1);
    }
    // qkv1 = x1 @ Wqkv1 ; qkv2 = x2 @ Wqkv2
    {
      GemmP p = mkGemm(x1p_bf, Dc, wq1b + (long long)l * Dc * 3 * Dc, 3 * Dc, ROWSc, 3 * Dc, Dc);
      p.Cbf = qkv1_bf; gemm(p, 1);
      GemmP q = mkGemm(x2p_bf, Dc, wq2b + (long long)l * Dc * 3 * Dc, 3 * Dc, ROWSc, 3 * Dc, Dc);
      q.Cbf = qkv2_bf; gemm(q, 1);
    }
    // vavg = (v1 + v2) / 2
    vavg_kernel<<<dim3(ROWSc * Dc / 256), dim3(256), 0, stream>>>(qkv1_bf, qkv2_bf, vavg_bf);

    // s1 = scale * Q1 K1^T ; s2 = scale * Q2 K2^T    (batched over B*H)
    {
      GemmP p = mkGemm(qkv1_bf, 3 * Dc, qkv1_bf + Dc, 3 * Dc, Sc, Sc, DHc);
      p.batchDiv = Hc; p.transB = 1; p.alpha = SCALEc;
      p.sA1 = (long long)Sc * 3 * Dc; p.sA2 = DHc;
      p.sB1 = (long long)Sc * 3 * Dc; p.sB2 = DHc;
      p.Cf = s1; p.ldc = Sc; p.sC1 = (long long)Hc * Sc * Sc; p.sC2 = (long long)Sc * Sc;
      gemm(p, Bc * Hc);
      p.A = qkv2_bf; p.B = qkv2_bf + Dc; p.Cf = s2;
      gemm(p, Bc * Hc);
    }
    // diff = softmax(s1) - softmax(s2)
    softmax_diff_kernel<<<dim3(Bc * Hc * Sc), dim3(256), 0, stream>>>(s1, s2, diffb);

    // attn_out = diff @ vavg   (batched; C scattered back to [B,S,H*DH])
    {
      GemmP p = mkGemm(diffb, Sc, vavg_bf, Dc, Sc, DHc, Sc);
      p.batchDiv = Hc;
      p.sA1 = (long long)Hc * Sc * Sc; p.sA2 = (long long)Sc * Sc;
      p.sB1 = (long long)Sc * Dc;      p.sB2 = DHc;
      p.Cbf = ao_bf; p.ldc = Dc;
      p.sC1 = (long long)Sc * Dc;      p.sC2 = DHc;
      gemm(p, Bc * Hc);
    }
    // tmp = attn_out @ Wo + bo ; xc += LN(tmp; g_out,b_out)
    {
      GemmP p = mkGemm(ao_bf, Dc, wob + (long long)l * Dc * Dc, Dc, ROWSc, Dc, Dc);
      p.bias = bo + l * Dc; p.Cf = tmpf; gemm(p, 1);
    }
    ln_kernel<<<dim3(ROWSc), dim3(256), 0, stream>>>(
        tmpf, g_out + l * Dc, b_out + l * Dc, nullptr, nullptr, nullptr, xc);

    // h2 = LN(xc; g2,b2) ; xc += gelu(h2 @ Wm1 + bm1) @ Wm2 + bm2
    ln_kernel<<<dim3(ROWSc), dim3(256), 0, stream>>>(
        xc, g2 + l * Dc, b2 + l * Dc, nullptr, nullptr, a_bf, nullptr);
    {
      GemmP p = mkGemm(a_bf, Dc, wm1b + (long long)l * Dc * 4 * Dc, 4 * Dc, ROWSc, 4 * Dc, Dc);
      p.bias = bm1 + (long long)l * 4 * Dc; p.gelu = 1; p.Cbf = m_bf; gemm(p, 1);
      GemmP q = mkGemm(m_bf, 4 * Dc, wm2b + (long long)l * 4 * Dc * Dc, Dc, ROWSc, Dc, 4 * Dc);
      q.bias = bm2 + l * Dc; q.Cf = xc; q.accumulate = 1; gemm(q, 1);
    }
  }

  // final LN -> d_out (f32)
  ln_kernel<<<dim3(ROWSc), dim3(256), 0, stream>>>(
      xc, gf, bfin, nullptr, (float*)d_out, nullptr, nullptr);
}
